// ConvIntrinsic_17102559772777
// MI455X (gfx1250) — matrix operand order repaired and checked by reading, not compiled, not run on hardware
//
#include <hip/hip_runtime.h>

// Problem constants (from reference): N_VERTS=50000, R=5, A=8, F=64, T=64
#define NVERTS 50000
#define NRAD   5
#define NANG   8
#define FEAT   64
#define TMPL   64
#define RA     40          // NRAD*NANG
#define CDIM   2624        // RA*FEAT + FEAT  (last 64 cols = mesh row for conv_center)
#define NDIM   512         // O(8) * TMPL(64)
#define KTILES 82          // CDIM / 32

typedef __attribute__((ext_vector_type(16))) __bf16 v16bf;
typedef __attribute__((ext_vector_type(8)))  __bf16 v8bf;
typedef __attribute__((ext_vector_type(8)))  float  v8f;

union BF16x16 { v16bf v; v8bf h[2]; };

// f32 -> bf16 bits, round-to-nearest-even (pure integer; no __bf16 scalar math)
static __device__ __forceinline__ unsigned f2bf(float f) {
    unsigned u = __builtin_bit_cast(unsigned, f);
    return (u + 0x7FFFu + ((u >> 16) & 1u)) >> 16;
}

// ---------------------------------------------------------------------------
// Kernel 0: gather + barycentric interp, pack A[k][c] (bf16, row-major).
//   c in [0,2560):    A[k][xy*64+f] = sum_j bw[k,xy,j] * mesh[idx[k,xy,j], f]
//   c in [2560,2624): A[k][2560+f]  = mesh[k, f]        (conv_center columns)
// One wave handles one (k, xy) pair; each lane packs 2 features into a u32.
// ---------------------------------------------------------------------------
__global__ __launch_bounds__(256) void pack_A_kernel(
    const float* __restrict__ mesh, const int* __restrict__ bidx,
    const float* __restrict__ bw, unsigned* __restrict__ Au)
{
    int gid  = blockIdx.x * 256 + threadIdx.x;
    int pair = gid >> 5;
    int lane = gid & 31;
    if (pair >= NVERTS * (RA + 1)) return;
    int k  = pair / (RA + 1);
    int xy = pair - k * (RA + 1);

    float a0, a1;
    int   col;
    if (xy < RA) {
        size_t base = ((size_t)k * RA + xy) * 3;
        a0 = 0.f; a1 = 0.f;
#pragma unroll
        for (int j = 0; j < 3; ++j) {
            int   id = bidx[base + j];
            float w  = bw[base + j];
            const float2* mrow = (const float2*)(mesh + (size_t)id * FEAT);
            float2 m = mrow[lane];
            a0 += w * m.x;
            a1 += w * m.y;
        }
        col = xy * 32 + lane;
    } else {
        const float2* mrow = (const float2*)(mesh + (size_t)k * FEAT);
        float2 m = mrow[lane];
        a0 = m.x; a1 = m.y;
        col = RA * 32 + lane;   // columns 2560..2623
    }
    Au[(size_t)k * (CDIM / 2) + col] = f2bf(a0) | (f2bf(a1) << 16);
}

// ---------------------------------------------------------------------------
// Kernel 1: fold 40x40 kernel matrix into template weights (0.1 GFLOP) and
// pack B column-major [N=512][C=2624] bf16.
//   n = o*64 + t
//   c < 2560 : B[n][xy*64+f] = sum_{r,a} nw[t,r,(a+o*rd)%8,f] * kern[r,a,x,y]
//   c >= 2560: B[n][2560+f]  = self_weights[t, f]
// ---------------------------------------------------------------------------
__global__ __launch_bounds__(256) void build_B_kernel(
    const float* __restrict__ nw, const float* __restrict__ kern,
    const float* __restrict__ sw, const int* __restrict__ rdp,
    unsigned short* __restrict__ Bh)
{
    int idx = blockIdx.x * 256 + threadIdx.x;
    if (idx >= NDIM * CDIM) return;
    int n = idx / CDIM;
    int c = idx - n * CDIM;
    int t = n & (TMPL - 1);
    int o = n >> 6;

    float acc;
    if (c < RA * FEAT) {
        int xy = c >> 6, f = c & 63;
        int x  = xy >> 3, y = xy & 7;
        int rd = rdp[0];
        acc = 0.f;
#pragma unroll
        for (int r = 0; r < NRAD; ++r)
#pragma unroll
            for (int a = 0; a < NANG; ++a) {
                int arot = (a + o * rd) & (NANG - 1);
                acc += nw[((t * NRAD + r) * NANG + arot) * FEAT + f] *
                       kern[((r * NANG + a) * NRAD + x) * NANG + y];
            }
    } else {
        acc = sw[t * FEAT + (c - RA * FEAT)];
    }
    Bh[(size_t)n * CDIM + c] = (unsigned short)f2bf(acc);
}

// ---------------------------------------------------------------------------
// Kernel 2: M=50000, N=512, K=2624 bf16 GEMM with fused bias + ReLU.
// Block = 8 waves covering a 32(M) x 512(N) patch: every wave owns
// 2 M-tiles x 4 N-tiles (8 f32 accumulators = 64 VGPRs), so each B fragment
// feeds two back-to-back WMMAs and per-k-step traffic is 12 b128 loads for
// 8 WMMAs (0.75 KB/WMMA) — halves L2 B-traffic vs 1Mx4N tiling.
// Fragment layouts (A/B: contiguous k order used by this code; D as in the ISA):
//   A (16x32, row-major src): lanes 0-15: K0..15 contiguous; lanes 16-31: K16..31
//   B (32x16, col-major src): lanes 0-15: K0..15 contiguous; lanes 16-31: K16..31
//   D (f32): lanes 0-15: M=v; lanes 16-31: M=8+v; N=lane&15
// Last block (rows 49984..50015): second tile's A rows clamp to 49999 for
// loads (EXEC stays all-ones for WMMA) and its stores are skipped uniformly.
// ---------------------------------------------------------------------------
__global__ __launch_bounds__(256) void gemm_bias_relu_kernel(
    const __bf16* __restrict__ A, const __bf16* __restrict__ B,
    const float* __restrict__ bias, float* __restrict__ out)
{
    const int  lane = threadIdx.x & 31;
    const int  wave = threadIdx.x >> 5;
    const int  ln   = lane & 15;
    const bool hi   = lane >= 16;
    const int  mt   = blockIdx.x;          // 32-row patch index

    const int row0 = mt * 32 + ln;         // M-tile 0 row for this lane
    int       row1 = mt * 32 + 16 + ln;    // M-tile 1 row (clamp OOB loads)
    if (row1 >= NVERTS) row1 = NVERTS - 1;

    const __bf16* arow0 = A + (size_t)row0 * CDIM;
    const __bf16* arow1 = A + (size_t)row1 * CDIM;
    const __bf16* bcol[4];
#pragma unroll
    for (int j = 0; j < 4; ++j)
        bcol[j] = B + (size_t)((wave * 4 + j) * 16 + ln) * CDIM;

    v8f acc0[4] = {};
    v8f acc1[4] = {};

    const int aoff0 = hi ? 16 : 0;
    const int aoff1 = hi ? 24 : 8;
    const int boff  = hi ? 16 : 0;

#pragma unroll 2
    for (int kc = 0; kc < KTILES; ++kc) {
        const int base = kc * 32;
        BF16x16 af0, af1;
        af0.h[0] = *(const v8bf*)(arow0 + base + aoff0);
        af0.h[1] = *(const v8bf*)(arow0 + base + aoff1);
        af1.h[0] = *(const v8bf*)(arow1 + base + aoff0);
        af1.h[1] = *(const v8bf*)(arow1 + base + aoff1);
#pragma unroll
        for (int j = 0; j < 4; ++j) {
            BF16x16 bfrag;
            bfrag.h[0] = *(const v8bf*)(bcol[j] + base + boff);
            bfrag.h[1] = *(const v8bf*)(bcol[j] + base + boff + 8);
            acc0[j] = __builtin_amdgcn_wmma_f32_16x16x32_bf16(
                false, af0.v, false, bfrag.v, (short)0, acc0[j], false, false);
            acc1[j] = __builtin_amdgcn_wmma_f32_16x16x32_bf16(
                false, af1.v, false, bfrag.v, (short)0, acc1[j], false, false);
        }
    }

    const int  mb0         = mt * 32 + (hi ? 8 : 0);
    const int  mb1         = mt * 32 + 16 + (hi ? 8 : 0);
    const bool tile1_valid = (mt * 32 + 16) < NVERTS;   // block-uniform

#pragma unroll
    for (int j = 0; j < 4; ++j) {
        const int   n  = (wave * 4 + j) * 16 + ln;
        const float bv = bias[n & (TMPL - 1)];
#pragma unroll
        for (int v = 0; v < 8; ++v) {
            float r = acc0[j][v] + bv;
            r = r > 0.f ? r : 0.f;
            out[(size_t)(mb0 + v) * NDIM + n] = r;
        }
        if (tile1_valid) {
#pragma unroll
            for (int v = 0; v < 8; ++v) {
                float r = acc1[j][v] + bv;
                r = r > 0.f ? r : 0.f;
                out[(size_t)(mb1 + v) * NDIM + n] = r;
            }
        }
    }
}

// ---------------------------------------------------------------------------
extern "C" void kernel_launch(void* const* d_in, const int* in_sizes, int n_in,
                              void* d_out, int out_size, void* d_ws, size_t ws_size,
                              hipStream_t stream)
{
    (void)in_sizes; (void)n_in; (void)out_size; (void)ws_size;

    const float* mesh = (const float*)d_in[0];
    const int*   bidx = (const int*)d_in[1];
    const float* bw   = (const float*)d_in[2];
    const float* kern = (const float*)d_in[3];
    const float* nw   = (const float*)d_in[4];
    const float* sw   = (const float*)d_in[5];
    const float* bias = (const float*)d_in[6];
    const int*   rdp  = (const int*)d_in[7];

    char* ws = (char*)d_ws;
    unsigned*       Au = (unsigned*)ws;                         // A: 50000*2624 bf16
    const size_t    A_bytes = (size_t)NVERTS * CDIM * 2;        // 262.4 MB
    unsigned short* Bh = (unsigned short*)(ws + A_bytes);       // B: 512*2624 bf16 (2.7 MB)

    {
        int total = NVERTS * (RA + 1) * 32;                     // one wave per (k,xy)
        pack_A_kernel<<<(total + 255) / 256, 256, 0, stream>>>(mesh, bidx, bw, Au);
    }
    {
        int total = NDIM * CDIM;
        build_B_kernel<<<(total + 255) / 256, 256, 0, stream>>>(nw, kern, sw, rdp, Bh);
    }
    gemm_bias_relu_kernel<<<(NVERTS + 31) / 32, 256, 0, stream>>>(
        (const __bf16*)Au, (const __bf16*)Bh, bias, (float*)d_out);
}
